// OuterProductMean_31516470018143
// MI455X (gfx1250) — matrix operand order repaired and checked by reading, not compiled, not run on hardware
//
#include <hip/hip_runtime.h>
#include <hip/hip_bf16.h>
#include <stdint.h>

// ---- WMMA fragment types (CDNA5, wave32) ----
typedef __attribute__((ext_vector_type(16))) __bf16        bf16x16; // 8 VGPRs
typedef __attribute__((ext_vector_type(8)))  float         f32x8;   // 8 VGPRs
typedef __attribute__((ext_vector_type(4)))  unsigned int  u32x4;   // 16B chunk

union Frag {
    u32x4   u4[2];
    bf16x16 v;
};

#define S_DIM 128
#define R_DIM 256
#define CM    256
#define CH    32
#define CZ    128
#define LN_EPS_F  1e-5f
#define MSK_EPS_F 1e-3f

// =====================================================================
// Kernel 1: LayerNorm over C_m + two 256->32 projections, mask applied.
// One wave per (s,r) row. Output layout a_t[r][c][s], b_t[r][c][s] (bf16)
// so stage-1 WMMA fragments are contiguous loads.
// =====================================================================
__global__ __launch_bounds__(256) void opm_prep_kernel(
    const float* __restrict__ m,        const float* __restrict__ mask,
    const float* __restrict__ ln_scale, const float* __restrict__ ln_bias,
    const float* __restrict__ w1,       const float* __restrict__ b1,
    const float* __restrict__ w2,       const float* __restrict__ b2,
    __bf16* __restrict__ a_t,           __bf16* __restrict__ b_t)
{
    __shared__ float lnLds[8][CM];      // 8 KB, one 256-row per wave
    const int wv   = threadIdx.x >> 5;
    const int lane = threadIdx.x & 31;
    const int row  = blockIdx.x * 8 + wv;   // row = s*R + r
    const int s    = row >> 8;              // R == 256
    const int r    = row & 255;
    const float* xrow = m + (size_t)row * CM;

    float x[8];
    float sum = 0.f, sq = 0.f;
    #pragma unroll
    for (int t = 0; t < 8; ++t) {
        x[t] = xrow[lane + 32 * t];
        sum += x[t];
        sq  += x[t] * x[t];
    }
    #pragma unroll
    for (int off = 1; off < 32; off <<= 1) {
        sum += __shfl_xor(sum, off, 32);
        sq  += __shfl_xor(sq,  off, 32);
    }
    const float mean = sum * (1.0f / CM);
    const float var  = sq  * (1.0f / CM) - mean * mean;   // biased var (jnp.var)
    const float rstd = rsqrtf(var + LN_EPS_F);
    #pragma unroll
    for (int t = 0; t < 8; ++t) {
        const int k = lane + 32 * t;
        lnLds[wv][k] = (x[t] - mean) * rstd * ln_scale[k] + ln_bias[k];
    }
    __syncthreads();

    // lane == output channel c (C_h == 32 == wave32)
    const float mk = mask[row];
    float av = b1[lane], bv = b2[lane];
    for (int k = 0; k < CM; ++k) {
        const float lv = lnLds[wv][k];
        av += lv * w1[k * CH + lane];   // coalesced across lanes
        bv += lv * w2[k * CH + lane];
    }
    av *= mk; bv *= mk;
    a_t[((size_t)r * CH + lane) * S_DIM + s] = (__bf16)av;
    b_t[((size_t)r * CH + lane) * S_DIM + s] = (__bf16)bv;
}

// =====================================================================
// Kernel 2: w_out [1024][128] f32 -> w_out_t [128][1024] bf16
// (makes stage-2 B fragments K-contiguous; 256 KB, L2-resident)
// =====================================================================
__global__ __launch_bounds__(256) void opm_wout_t_kernel(
    const float* __restrict__ w_out, __bf16* __restrict__ w_out_t)
{
    const int idx = blockIdx.x * 256 + threadIdx.x;   // over 1024*128
    const int k = idx >> 7;
    const int z = idx & 127;
    w_out_t[(size_t)z * 1024 + k] = (__bf16)w_out[idx];
}

// =====================================================================
// Kernel 3 (fused): per 4x4 (i,j) tile:
//   stage 1: outer[c,e] = sum_s a[i,s,c] b[j,s,e]   (32x32x128 GEMM, bf16 WMMA)
//   stage 2: out[i,j,z] = (vec(outer) @ w_out + b_out) / (mask.mask + eps)
// 8 waves / 256 threads, 96 KB LDS (WGP has 320 KB).
// =====================================================================
__global__ __launch_bounds__(256) void opm_main_kernel(
    const __bf16* __restrict__ a_t, const __bf16* __restrict__ b_t,
    const __bf16* __restrict__ w_out_t,
    const float* __restrict__ mask, const float* __restrict__ b_out,
    float* __restrict__ out)
{
    __shared__ __bf16 aLds[4 * CH * S_DIM];   // 32 KB : 4 i-rows
    __shared__ __bf16 bLds[4 * CH * S_DIM];   // 32 KB : 4 j-rows
    __shared__ __bf16 outerLds[16 * CH * CH]; // 32 KB : 16 pairs x 1024 (bf16)
    __shared__ float  rnormLds[16];

    const int tid = threadIdx.x;
    const int jb  = blockIdx.x;   // j-tile (fast) -> coalesced output
    const int ib  = blockIdx.y;   // i-tile

    // cooperative 16B-chunk fills of a/b tiles
    {
        const u32x4* ga = (const u32x4*)(a_t + (size_t)ib * 4 * CH * S_DIM);
        const u32x4* gb = (const u32x4*)(b_t + (size_t)jb * 4 * CH * S_DIM);
        u32x4* la = (u32x4*)aLds;
        u32x4* lb = (u32x4*)bLds;
        #pragma unroll
        for (int t = 0; t < 8; ++t) {              // 2048 chunks / 256 threads
            la[t * 256 + tid] = ga[t * 256 + tid];
            lb[t * 256 + tid] = gb[t * 256 + tid];
        }
    }
    // mask normalization: norm(i,j) = sum_s mask[s,i]*mask[s,j] + eps
    if (tid < 16) {
        const int i = ib * 4 + (tid >> 2);
        const int j = jb * 4 + (tid & 3);
        float sum = 0.f;
        for (int s = 0; s < S_DIM; ++s)
            sum += mask[s * R_DIM + i] * mask[s * R_DIM + j];
        rnormLds[tid] = 1.0f / (sum + MSK_EPS_F);
    }
    __syncthreads();

    const int w  = tid >> 5;       // wave 0..7
    const int l  = tid & 31;       // lane
    const int lo = l & 15;         // M/N index within fragment
    const int hi = l >> 4;         // lane-group select

    // hint the L2-resident weight rows for stage 2
    __builtin_prefetch(w_out_t + (size_t)(w * 16 + lo) * 1024, 0, 1);

    // ---------- stage 1: wave handles pairs 2w and 2w+1 ----------
    #pragma unroll
    for (int pp = 0; pp < 2; ++pp) {
        const int p  = w * 2 + pp;
        const int il = p >> 2, jl = p & 3;
        f32x8 acc00 = {}, acc01 = {}, acc10 = {}, acc11 = {};
        #pragma unroll
        for (int k = 0; k < 4; ++k) {             // K = 128 = 4 x 32
            Frag A0, A1, B0, B1;
            // A (16x32 bf16): lane row m=lo, K runs {klo..klo+7, klo+16..klo+23}
            const __bf16* pa0 = &aLds[(il * CH +      lo) * S_DIM + k * 32 + hi * 8];
            const __bf16* pa1 = &aLds[(il * CH + 16 + lo) * S_DIM + k * 32 + hi * 8];
            A0.u4[0] = *(const u32x4*)pa0;  A0.u4[1] = *(const u32x4*)(pa0 + 16);
            A1.u4[0] = *(const u32x4*)pa1;  A1.u4[1] = *(const u32x4*)(pa1 + 16);
            // B (32x16 bf16): lane col n=lo, K runs {klo..klo+7, klo+16..klo+23}
            const __bf16* pb0 = &bLds[(jl * CH +      lo) * S_DIM + k * 32 + hi * 8];
            const __bf16* pb1 = &bLds[(jl * CH + 16 + lo) * S_DIM + k * 32 + hi * 8];
            B0.u4[0] = *(const u32x4*)pb0;  B0.u4[1] = *(const u32x4*)(pb0 + 16);
            B1.u4[0] = *(const u32x4*)pb1;  B1.u4[1] = *(const u32x4*)(pb1 + 16);

            acc00 = __builtin_amdgcn_wmma_f32_16x16x32_bf16(false, A0.v, false, B0.v, (short)0, acc00, false, false);
            acc01 = __builtin_amdgcn_wmma_f32_16x16x32_bf16(false, A0.v, false, B1.v, (short)0, acc01, false, false);
            acc10 = __builtin_amdgcn_wmma_f32_16x16x32_bf16(false, A1.v, false, B0.v, (short)0, acc10, false, false);
            acc11 = __builtin_amdgcn_wmma_f32_16x16x32_bf16(false, A1.v, false, B1.v, (short)0, acc11, false, false);
        }
        // scatter outer (bf16) into LDS as [p][c*32+e] (= A rows for stage 2)
        __bf16* od = &outerLds[p * CH * CH];
        #pragma unroll
        for (int v = 0; v < 8; ++v) {
            const int c = v + 8 * hi;             // C/D layout: m = v + 8*(l>>4)
            od[(c)      * CH + lo     ] = (__bf16)acc00[v];
            od[(c)      * CH + lo + 16] = (__bf16)acc01[v];
            od[(c + 16) * CH + lo     ] = (__bf16)acc10[v];
            od[(c + 16) * CH + lo + 16] = (__bf16)acc11[v];
        }
    }
    __syncthreads();

    // ---------- stage 2: [16 pairs x 1024] @ [1024 x 128], wave w -> z-tile w ----------
    f32x8 acc = {};
    const __bf16* wrow = w_out_t + (size_t)(w * 16 + lo) * 1024; // z = 16w+lo row, K-contig
    #pragma unroll 4
    for (int step = 0; step < 32; ++step) {       // K = 1024 = 32 x 32
        Frag A, B;
        const __bf16* pA = &outerLds[lo * 1024 + step * 32 + hi * 8];
        A.u4[0] = *(const u32x4*)pA;  A.u4[1] = *(const u32x4*)(pA + 16);
        const __bf16* pB = wrow + step * 32 + hi * 8;
        B.u4[0] = *(const u32x4*)pB;  B.u4[1] = *(const u32x4*)(pB + 16);
        acc = __builtin_amdgcn_wmma_f32_16x16x32_bf16(false, A.v, false, B.v, (short)0, acc, false, false);
    }
    // epilogue: D tile m = pair p, n = z offset
    const int   z  = w * 16 + lo;
    const float bz = b_out[z];
    #pragma unroll
    for (int v = 0; v < 8; ++v) {
        const int p = v + 8 * hi;
        const int i = ib * 4 + (p >> 2);
        const int j = jb * 4 + (p & 3);
        out[((size_t)i * R_DIM + j) * CZ + z] = (acc[v] + bz) * rnormLds[p];
    }
}

// =====================================================================
// Launch: inputs per setup_inputs() order:
// 0:m 1:mask 2:ln_scale 3:ln_bias 4:w1 5:b1 6:w2 7:b2 8:w_out 9:b_out
// Workspace: a_t (2MB) | b_t (2MB) | w_out_t (256KB)  -> 4.25 MB needed
// =====================================================================
extern "C" void kernel_launch(void* const* d_in, const int* in_sizes, int n_in,
                              void* d_out, int out_size, void* d_ws, size_t ws_size,
                              hipStream_t stream) {
    const float* m        = (const float*)d_in[0];
    const float* mask     = (const float*)d_in[1];
    const float* ln_scale = (const float*)d_in[2];
    const float* ln_bias  = (const float*)d_in[3];
    const float* w1       = (const float*)d_in[4];
    const float* b1       = (const float*)d_in[5];
    const float* w2       = (const float*)d_in[6];
    const float* b2       = (const float*)d_in[7];
    const float* w_out    = (const float*)d_in[8];
    const float* b_out    = (const float*)d_in[9];
    float* out = (float*)d_out;

    __bf16* a_t     = (__bf16*)d_ws;
    __bf16* b_t     = a_t + (size_t)R_DIM * CH * S_DIM;      // +1,048,576 elems
    __bf16* w_out_t = b_t + (size_t)R_DIM * CH * S_DIM;      // +1,048,576 elems

    // 32768 rows, 8 per block
    opm_prep_kernel<<<4096, 256, 0, stream>>>(m, mask, ln_scale, ln_bias,
                                              w1, b1, w2, b2, a_t, b_t);
    // 1024*128 elements, 256 per block
    opm_wout_t_kernel<<<512, 256, 0, stream>>>(w_out, w_out_t);
    // (R/4) x (R/4) pair tiles
    opm_main_kernel<<<dim3(64, 64), 256, 0, stream>>>(a_t, b_t, w_out_t,
                                                      mask, b_out, out);
}
